// GRU_43550968382065
// MI455X (gfx1250) — matrix operand order repaired and checked by reading, not compiled, not run on hardware
//
#include <hip/hip_runtime.h>

// GRU (Haste variant) for MI455X / gfx1250, wave32 + WMMA bf16.
// T=2048, B=64, I=256, H=256, gates [z|r|g] along 3H=768.

typedef __attribute__((ext_vector_type(16))) __bf16 v16bf;
typedef __attribute__((ext_vector_type(8)))  float  v8f;

#define T_STEPS 2048
#define BATCH   64
#define HID     256
#define NWG     8

union BF16x16 {
    v16bf        v;
    unsigned int u[8];
};

__device__ __forceinline__ unsigned int cvt2(float lo, float hi) {
    union { __bf16 h[2]; unsigned int u; } r;
    r.h[0] = (__bf16)lo; r.h[1] = (__bf16)hi;
    return r.u;
}
__device__ __forceinline__ unsigned short cvt1(float f) {
    union { __bf16 h; unsigned short s; } r;
    r.h = (__bf16)f; return r.s;
}
__device__ __forceinline__ float sigmoidf_(float x) {
    return 1.0f / (1.0f + __expf(-x));
}

// ---------------------------------------------------------------------------
// Init: zero h0 output row, zero h-exchange buffer 0, reset step counter.
// ---------------------------------------------------------------------------
__global__ void gru_init(float* __restrict__ out,
                         unsigned short* __restrict__ hbuf,
                         unsigned int* __restrict__ counter) {
    int i = blockIdx.x * blockDim.x + threadIdx.x;
    if (i < BATCH * HID) {
        out[i]  = 0.0f;
        hbuf[i] = 0;
    }
    if (i == 0) *counter = 0u;
}

// ---------------------------------------------------------------------------
// One-shot f32 -> bf16 conversion of x (memory-bound; ~200 MB of traffic).
// Row-major layout preserved so the scan's A-operand loads are b128s.
// ---------------------------------------------------------------------------
__global__ void __launch_bounds__(256, 1)
xconv(const float* __restrict__ x, unsigned short* __restrict__ xb) {
    const size_t i = ((size_t)blockIdx.x * 256 + threadIdx.x) * 8;
    const float4 f0 = *reinterpret_cast<const float4*>(x + i);
    const float4 f1 = *reinterpret_cast<const float4*>(x + i + 4);
    uint4 o;
    o.x = cvt2(f0.x, f0.y);
    o.y = cvt2(f0.z, f0.w);
    o.z = cvt2(f1.x, f1.y);
    o.w = cvt2(f1.z, f1.w);
    *reinterpret_cast<uint4*>(xb + i) = o;
}

// ---------------------------------------------------------------------------
// Persistent GRU scan. Grid = 8 blocks x 256 threads (8 waves).
// Block wg owns hidden units [wg*32, wg*32+32). Weight slice [kernel;R]
// (512 x 96 cols: z|r|g) lives in LDS in bf16, pre-swizzled to the WMMA
// B-operand lane layout. Wave (mb,ch): batch tile mb, column half ch.
// Per step: 48 x v_wmma_f32_16x16x32_bf16 -> gate math -> device-wide
// step barrier (atomic counter + s_sleep spin), bf16 h exchange via L2.
// ---------------------------------------------------------------------------
template <bool XBF>
__global__ void __launch_bounds__(256, 1)
gru_scan(const float* __restrict__ x,            // (T, B, 256) f32
         const unsigned short* __restrict__ xb,  // (T, B, 256) bf16 (XBF path)
         const float* __restrict__ Wk,           // (256, 768)
         const float* __restrict__ Rk,           // (256, 768)
         const float* __restrict__ bias,         // (768,)
         const float* __restrict__ rbias,        // (768,)
         float* __restrict__ out,                // (T+1, B, H) then (1, B, H)
         unsigned short* __restrict__ hbuf,      // [2][B*H] bf16 exchange
         unsigned int* __restrict__ counter)
{
    extern __shared__ unsigned int ldsB[];   // 6 * 16 * 32 * 8 dwords = 96 KB

    const int tid  = threadIdx.x;
    const int lane = tid & 31;
    const int wave = tid >> 5;
    const int wg   = blockIdx.x;     // hidden slice 0..7
    const int mb   = wave & 3;       // batch tile 0..3
    const int ch   = wave >> 2;      // column half 0..1

    // ---- Stage weight slice into LDS, pre-swizzled to B-operand layout ----
    // B (16-bit, 32x16): lane holds col n=lane&15, K group (lane>>4)*8,
    // dword j holds K = group+2j(+8 if j>=4), and the next K.
    for (int idx = wave; idx < 96; idx += 8) {
        const int nt   = idx >> 4;
        const int kc   = idx & 15;
        const int gate = nt >> 1;
        const int col  = gate * HID + wg * 32 + (nt & 1) * 16 + (lane & 15);
        const int kg   = (lane >> 4) * 8;
        unsigned int* dst = &ldsB[(idx * 32 + lane) * 8];
#pragma unroll
        for (int j = 0; j < 8; ++j) {
            const int k0 = kc * 32 + kg + 2 * j + (j >= 4 ? 8 : 0);
            const int k1 = k0 + 1;
            const float lo = (k0 < 256) ? Wk[k0 * 768 + col] : Rk[(k0 - 256) * 768 + col];
            const float hi = (k1 < 256) ? Wk[k1 * 768 + col] : Rk[(k1 - 256) * 768 + col];
            dst[j] = cvt2(lo, hi);
        }
    }
    __syncthreads();

    // ---- Per-lane constants ----
    const int n    = lane & 15;
    const int colL = wg * 32 + ch * 16 + n;
    const float bz  = bias[colL]           + rbias[colL];
    const float br  = bias[HID + colL]     + rbias[HID + colL];
    const float bgx = bias[2 * HID + colL];
    const float bgh = rbias[2 * HID + colL];

    const int ma  = mb * 16 + (lane & 15);        // A-operand row (batch)
    const int mo  = mb * 16 + (lane >> 4) * 8;    // C-layout row base
    const int kgA = (lane >> 4) * 8;              // A-operand K sub-offset

    const unsigned int* lbz = &ldsB[((0 + ch) * 16) * 32 * 8];
    const unsigned int* lbr = &ldsB[((2 + ch) * 16) * 32 * 8];
    const unsigned int* lbg = &ldsB[((4 + ch) * 16) * 32 * 8];

    float hprev[8];
#pragma unroll
    for (int v = 0; v < 8; ++v) hprev[v] = 0.0f;

    for (int t = 0; t < T_STEPS; ++t) {
        v8f acc_z = {}, acc_r = {}, acc_gx = {}, acc_gh = {};

        const float*          xrowf = x  + (size_t)t * (BATCH * 256) + ma * 256 + kgA;
        const unsigned short* xrowb = xb + (size_t)t * (BATCH * 256) + ma * 256 + kgA;
        const unsigned short* hrow  = hbuf + (size_t)(t & 1) * (BATCH * HID) + ma * HID + kgA;

        // Prefetch next timestep's x rows (global_prefetch_b8) while this
        // step's dependent GEMM runs.
        if (t + 1 < T_STEPS) {
            if (XBF) __builtin_prefetch(xrowb + BATCH * 256, 0, 3);
            else     __builtin_prefetch(xrowf + BATCH * 256, 0, 3);
        }

#pragma unroll
        for (int kc = 0; kc < 16; ++kc) {
            // ---- A operand (16x32 bf16): rows=batch, K over [x_t | h] ----
            BF16x16 a;
            if (kc < 8) {
                if (XBF) {
                    const unsigned short* xp = xrowb + kc * 32;
                    const uint4 q0 = *reinterpret_cast<const uint4*>(xp);
                    const uint4 q1 = *reinterpret_cast<const uint4*>(xp + 16);
                    a.u[0] = q0.x; a.u[1] = q0.y; a.u[2] = q0.z; a.u[3] = q0.w;
                    a.u[4] = q1.x; a.u[5] = q1.y; a.u[6] = q1.z; a.u[7] = q1.w;
                } else {
                    const float* p = xrowf + kc * 32;
                    const float4 f0 = *reinterpret_cast<const float4*>(p);
                    const float4 f1 = *reinterpret_cast<const float4*>(p + 4);
                    const float4 f2 = *reinterpret_cast<const float4*>(p + 16);
                    const float4 f3 = *reinterpret_cast<const float4*>(p + 20);
                    a.u[0] = cvt2(f0.x, f0.y); a.u[1] = cvt2(f0.z, f0.w);
                    a.u[2] = cvt2(f1.x, f1.y); a.u[3] = cvt2(f1.z, f1.w);
                    a.u[4] = cvt2(f2.x, f2.y); a.u[5] = cvt2(f2.z, f2.w);
                    a.u[6] = cvt2(f3.x, f3.y); a.u[7] = cvt2(f3.z, f3.w);
                }
            } else {
                const unsigned short* hp = hrow + (kc - 8) * 32;
                const uint4 q0 = *reinterpret_cast<const uint4*>(hp);
                const uint4 q1 = *reinterpret_cast<const uint4*>(hp + 16);
                a.u[0] = q0.x; a.u[1] = q0.y; a.u[2] = q0.z; a.u[3] = q0.w;
                a.u[4] = q1.x; a.u[5] = q1.y; a.u[6] = q1.z; a.u[7] = q1.w;
            }

            // ---- B operands: 2 x ds_load_b128 each from pre-swizzled LDS ----
            const uint4* pz = reinterpret_cast<const uint4*>(lbz + (kc * 32 + lane) * 8);
            const uint4* pr = reinterpret_cast<const uint4*>(lbr + (kc * 32 + lane) * 8);
            const uint4* pg = reinterpret_cast<const uint4*>(lbg + (kc * 32 + lane) * 8);
            BF16x16 bz16, br16, bg16;
            {
                const uint4 z0 = pz[0], z1 = pz[1];
                const uint4 r0 = pr[0], r1 = pr[1];
                const uint4 g0 = pg[0], g1 = pg[1];
                bz16.u[0] = z0.x; bz16.u[1] = z0.y; bz16.u[2] = z0.z; bz16.u[3] = z0.w;
                bz16.u[4] = z1.x; bz16.u[5] = z1.y; bz16.u[6] = z1.z; bz16.u[7] = z1.w;
                br16.u[0] = r0.x; br16.u[1] = r0.y; br16.u[2] = r0.z; br16.u[3] = r0.w;
                br16.u[4] = r1.x; br16.u[5] = r1.y; br16.u[6] = r1.z; br16.u[7] = r1.w;
                bg16.u[0] = g0.x; bg16.u[1] = g0.y; bg16.u[2] = g0.z; bg16.u[3] = g0.w;
                bg16.u[4] = g1.x; bg16.u[5] = g1.y; bg16.u[6] = g1.z; bg16.u[7] = g1.w;
            }

            acc_z = __builtin_amdgcn_wmma_f32_16x16x32_bf16(false, a.v, false, bz16.v,
                                                            (short)0, acc_z, false, false);
            acc_r = __builtin_amdgcn_wmma_f32_16x16x32_bf16(false, a.v, false, br16.v,
                                                            (short)0, acc_r, false, false);
            if (kc < 8)
                acc_gx = __builtin_amdgcn_wmma_f32_16x16x32_bf16(false, a.v, false, bg16.v,
                                                                 (short)0, acc_gx, false, false);
            else
                acc_gh = __builtin_amdgcn_wmma_f32_16x16x32_bf16(false, a.v, false, bg16.v,
                                                                 (short)0, acc_gh, false, false);
        }

        // ---- Gates + state update (C layout: dword v -> row mo+v, col colL) ----
        float* orow = out + (size_t)(t + 1) * (BATCH * HID);
        unsigned short* hw = hbuf + (size_t)((t + 1) & 1) * (BATCH * HID);
#pragma unroll
        for (int v = 0; v < 8; ++v) {
            const float z  = sigmoidf_(acc_z[v] + bz);
            const float r  = sigmoidf_(acc_r[v] + br);
            const float g  = tanhf(acc_gx[v] + bgx + r * (acc_gh[v] + bgh));
            const float hn = z * hprev[v] + (1.0f - z) * g;
            hprev[v] = hn;
            const int m = mo + v;
            orow[m * HID + colL] = hn;
            hw[m * HID + colL]   = cvt1(hn);
        }

        // ---- Device-wide step barrier across the 8 persistent WGs ----
        __threadfence();
        __syncthreads();
        if (tid == 0) {
            atomicAdd(counter, 1u);
            const unsigned int target = (unsigned int)NWG * (unsigned int)(t + 1);
            while (atomicAdd(counter, 0u) < target) __builtin_amdgcn_s_sleep(1);
        }
        __syncthreads();
        __threadfence();
    }

    // ---- Final state tuple element: h[T] ----
    float* srow = out + (size_t)(T_STEPS + 1) * (BATCH * HID);
#pragma unroll
    for (int v = 0; v < 8; ++v) {
        srow[(mo + v) * HID + colL] = hprev[v];
    }
}

// ---------------------------------------------------------------------------
extern "C" void kernel_launch(void* const* d_in, const int* in_sizes, int n_in,
                              void* d_out, int out_size, void* d_ws, size_t ws_size,
                              hipStream_t stream) {
    const float* x     = (const float*)d_in[0];
    const float* Wk    = (const float*)d_in[1];
    const float* Rk    = (const float*)d_in[2];
    const float* bias  = (const float*)d_in[3];
    const float* rbias = (const float*)d_in[4];
    float* out = (float*)d_out;

    // Workspace layout:
    //   [0..127]            step counter (padded)
    //   [128..128+64K)      bf16 h double buffer (2 x B x H)
    //   [aligned..+67MB)    bf16 copy of x (only if ws is big enough)
    unsigned int*   counter = (unsigned int*)d_ws;
    unsigned short* hbuf    = (unsigned short*)((char*)d_ws + 128);

    const size_t xoff   = (128 + 2 * (size_t)BATCH * HID * sizeof(unsigned short) + 255) & ~(size_t)255;
    const size_t xbytes = (size_t)T_STEPS * BATCH * 256 * sizeof(unsigned short);
    unsigned short* xbf = (unsigned short*)((char*)d_ws + xoff);
    const bool bigWs = (ws_size >= xoff + xbytes);

    gru_init<<<dim3((BATCH * HID + 255) / 256), dim3(256), 0, stream>>>(out, hbuf, counter);

    const size_t ldsBytes = 6 * 16 * 32 * 8 * sizeof(unsigned int);  // 96 KB

    if (bigWs) {
        const int nconv = (int)(((size_t)T_STEPS * BATCH * 256) / (8 * 256));
        xconv<<<dim3(nconv), dim3(256), 0, stream>>>(x, xbf);
        gru_scan<true><<<dim3(NWG), dim3(256), ldsBytes, stream>>>(
            x, xbf, Wk, Rk, bias, rbias, out, hbuf, counter);
    } else {
        gru_scan<false><<<dim3(NWG), dim3(256), ldsBytes, stream>>>(
            x, xbf, Wk, Rk, bias, rbias, out, hbuf, counter);
    }
}
